// AttentionBlock_25718264168858
// MI455X (gfx1250) — matrix operand order repaired and checked by reading, not compiled, not run on hardware
//
#include <hip/hip_runtime.h>
#include <math.h>

typedef __attribute__((ext_vector_type(16))) _Float16 v16h;
typedef __attribute__((ext_vector_type(8)))  _Float16 v8h;
typedef __attribute__((ext_vector_type(8)))  float    v8f;

#define C_DIM 512
#define N_TOK 4096   // 64*64 spatial positions
#define BATCH 2
#define GROUPS 32
#define CG 16        // channels per group

// ---------------- WMMA helpers (gfx1250, wave32) ----------------

__device__ __forceinline__ v16h make_v16(v8h lo, v8h hi) {
  v16h r;
#pragma unroll
  for (int i = 0; i < 8; ++i) { r[i] = lo[i]; r[i + 8] = hi[i]; }
  return r;
}

// A fragment: 16x32 f16 tile from row-major matrix, leading dim `ld` (halves).
// Lanes 0-15: rows 0..15, K 0-7 / 16-23; lanes 16-31: K 8-15 / 24-31.
__device__ __forceinline__ v16h load_a16(const _Float16* base, int row0, int k0, int ld) {
  int lane = threadIdx.x & 31;
  const _Float16* p = base + (size_t)(row0 + (lane & 15)) * ld + k0 + ((lane >> 4) << 3);
  v8h lo = *(const v8h*)p;
  v8h hi = *(const v8h*)(p + 16);
  return make_v16(lo, hi);
}

// B fragment: 32x16 f16 (KxN) from row-major KxN matrix, leading dim `ld`.
// Lanes 0-15: cols 0..15, K 0-7 / 16-23; lanes 16-31: K 8-15 / 24-31.
__device__ __forceinline__ v16h load_b16(const _Float16* base, int k0, int n0, int ld) {
  int lane = threadIdx.x & 31;
  const _Float16* p = base + (size_t)(k0 + ((lane >> 4) << 3)) * ld + n0 + (lane & 15);
  v16h r;
#pragma unroll
  for (int i = 0; i < 8; ++i) { r[i] = p[(size_t)i * ld]; r[i + 8] = p[(size_t)(i + 16) * ld]; }
  return r;
}

__device__ __forceinline__ v8f wmma_f16(v16h a, v16h b, v8f c) {
  return __builtin_amdgcn_wmma_f32_16x16x32_f16(false, a, false, b, (short)0, c, false, false);
}

// ---------------- CDNA5 async global->LDS copy (ASYNCcnt-tracked) ----------------

__device__ __forceinline__ void async_copy_b128(unsigned lds_byte_off, const void* g) {
  asm volatile("global_load_async_to_lds_b128 %0, %1, off"
               :: "v"(lds_byte_off), "v"((unsigned long long)(uintptr_t)g)
               : "memory");
}
__device__ __forceinline__ void wait_async0() {
  asm volatile("s_wait_asynccnt 0x0" ::: "memory");
}

// ---------------- small utility kernels ----------------

__global__ void f32_to_f16_kernel(const float* __restrict__ src, _Float16* __restrict__ dst, int n) {
  int i = blockIdx.x * blockDim.x + threadIdx.x;
  int stride = gridDim.x * blockDim.x;
  for (; i < n; i += stride) dst[i] = (_Float16)src[i];
}

__global__ void gn_stats_kernel(const float* __restrict__ x, float2* __restrict__ stats) {
  int bg = blockIdx.x;               // b*GROUPS + g
  int b = bg >> 5, g = bg & 31;
  const float* xb = x + (size_t)b * N_TOK * C_DIM + g * CG;
  float s = 0.f, ss = 0.f;
  for (int i = threadIdx.x; i < N_TOK * CG; i += blockDim.x) {
    int n = i >> 4, c = i & 15;
    float vv = xb[(size_t)n * C_DIM + c];
    s += vv; ss += vv * vv;
  }
  __shared__ float rs[256], rss[256];
  rs[threadIdx.x] = s; rss[threadIdx.x] = ss;
  __syncthreads();
  for (int off = 128; off > 0; off >>= 1) {
    if ((int)threadIdx.x < off) { rs[threadIdx.x] += rs[threadIdx.x + off]; rss[threadIdx.x] += rss[threadIdx.x + off]; }
    __syncthreads();
  }
  if (threadIdx.x == 0) {
    const float inv = 1.0f / (float)(N_TOK * CG);
    float mean = rs[0] * inv;
    float var  = rss[0] * inv - mean * mean;
    stats[bg] = make_float2(mean, rsqrtf(var + 1e-5f));
  }
}

__global__ void gn_apply_kernel(const float* __restrict__ x, const float* __restrict__ gamma,
                                const float* __restrict__ beta, const float2* __restrict__ stats,
                                _Float16* __restrict__ xn) {
  size_t i = (size_t)blockIdx.x * blockDim.x + threadIdx.x;
  const size_t total = (size_t)BATCH * N_TOK * C_DIM;
  const size_t stride = (size_t)gridDim.x * blockDim.x;
  for (; i < total; i += stride) {
    int c = (int)(i & (C_DIM - 1));
    int b = (int)(i >> 21);          // N_TOK*C_DIM == 2^21
    float2 st = stats[b * GROUPS + (c >> 4)];
    xn[i] = (_Float16)((x[i] - st.x) * st.y * gamma[c] + beta[c]);
  }
}

// ---------------- WMMA GEMM, software-pipelined K loop ----------------
// MODE 0: out16 = (A@W + bias) * alpha   (Q/K/V)
// MODE 1: out32 = A@W + bias + residual  (projection)
template <int MODE>
__global__ void __launch_bounds__(256)
gemm_kernel(const _Float16* __restrict__ A, const _Float16* __restrict__ W,
            const float* __restrict__ bias, float alpha,
            _Float16* __restrict__ out16,
            const float* __restrict__ residual, float* __restrict__ out32) {
  int wave = threadIdx.x >> 5;
  int lane = threadIdx.x & 31;
  int rowg = wave >> 1, colg = wave & 1;
  int row0 = blockIdx.x * 64 + rowg * 16;
  int col0 = blockIdx.y * 128 + colg * 64;

  v8f acc[4] = {};
  v16h a  = load_a16(A, row0, 0, C_DIM);
  v16h b0 = load_b16(W, 0, col0,      C_DIM);
  v16h b1 = load_b16(W, 0, col0 + 16, C_DIM);
  v16h b2 = load_b16(W, 0, col0 + 32, C_DIM);
  v16h b3 = load_b16(W, 0, col0 + 48, C_DIM);
  for (int k0 = 0; k0 < C_DIM - 32; k0 += 32) {
    v16h an  = load_a16(A, row0, k0 + 32, C_DIM);
    v16h bn0 = load_b16(W, k0 + 32, col0,      C_DIM);
    v16h bn1 = load_b16(W, k0 + 32, col0 + 16, C_DIM);
    v16h bn2 = load_b16(W, k0 + 32, col0 + 32, C_DIM);
    v16h bn3 = load_b16(W, k0 + 32, col0 + 48, C_DIM);
    acc[0] = wmma_f16(a, b0, acc[0]);
    acc[1] = wmma_f16(a, b1, acc[1]);
    acc[2] = wmma_f16(a, b2, acc[2]);
    acc[3] = wmma_f16(a, b3, acc[3]);
    a = an; b0 = bn0; b1 = bn1; b2 = bn2; b3 = bn3;
  }
  acc[0] = wmma_f16(a, b0, acc[0]);
  acc[1] = wmma_f16(a, b1, acc[1]);
  acc[2] = wmma_f16(a, b2, acc[2]);
  acc[3] = wmma_f16(a, b3, acc[3]);

  int rbase = row0 + ((lane >> 4) << 3);
  int cl = lane & 15;
#pragma unroll
  for (int t = 0; t < 4; ++t) {
    int col = col0 + t * 16 + cl;
    float bv = bias[col];
#pragma unroll
    for (int r = 0; r < 8; ++r) {
      size_t idx = (size_t)(rbase + r) * C_DIM + col;
      float vres = acc[t][r] + bv;
      if (MODE == 0) out16[idx] = (_Float16)(vres * alpha);
      else           out32[idx] = vres + residual[idx];
    }
  }
}

// ---------------- fused flash-style attention ----------------
// Block = 256 threads (8 wave32), owns BR=64 query rows.
// Scores computed transposed (S^T = K · Q^T) so K needs NO transpose staging:
// A-fragments stream from row-major global k16; Q^T is transposed into LDS once.
// V chunks are staged into LDS with CDNA5 async global->LDS copies, issued at
// chunk start and waited only before the P·V WMMAs (overlaps score+softmax).
#define BR 64
#define BC 32
#define QT_LD 72    // padded ld (halves): 144B rows -> 16 distinct LDS banks
#define SV_LD 520   // padded ld (halves): 1040B rows
#define S_LD  36    // padded ld (floats)
#define P_LD  40    // padded ld (halves)

__global__ void __launch_bounds__(256)
attention_kernel(const _Float16* __restrict__ q, const _Float16* __restrict__ k,
                 const _Float16* __restrict__ v, _Float16* __restrict__ o) {
  __shared__ __align__(32) _Float16 qT[C_DIM * QT_LD];   // Q^T tile  (73.7 KB)
  __shared__ __align__(32) _Float16 sV[BC * SV_LD];      // V chunk   (33.3 KB)
  __shared__ __align__(32) float    sS[BR * S_LD];       // scores    (9.2 KB)
  __shared__ __align__(32) _Float16 sP[BR * P_LD];       // exp(s-m)  (5.1 KB)
  __shared__ float mrow[BR], lrow[BR], frow[BR];

  int b = blockIdx.y;
  int tid = threadIdx.x;
  int wave = tid >> 5, lane = tid & 31;
  int rowg = wave >> 1;   // 0..3 : 16 query rows each
  int colg = wave & 1;    // score: key-group; AV: 256-channel half

  const _Float16* qb = q + ((size_t)b * N_TOK + blockIdx.x * BR) * C_DIM;
  const _Float16* kb = k + (size_t)b * N_TOK * C_DIM;
  const _Float16* vb = v + (size_t)b * N_TOK * C_DIM;

  // --- one-time: transpose the Q tile into LDS qT[ch][row] ---
  {
    int row = tid & 63;
    int cg  = (tid >> 6) * 128;
#pragma unroll
    for (int j = 0; j < 16; ++j) {
      v8h d = *(const v8h*)(qb + (size_t)row * C_DIM + cg + j * 8);
#pragma unroll
      for (int i = 0; i < 8; ++i)
        qT[(size_t)(cg + j * 8 + i) * QT_LD + row] = d[i];
    }
  }
  if (tid < BR) { mrow[tid] = -1e30f; lrow[tid] = 0.f; }
  __syncthreads();

  v8f acc[16] = {};   // wave's O tile: 16 rows x 256 cols, f32

  const unsigned svbase = (unsigned)(uintptr_t)sV;
  const int skey = tid >> 3;          // 0..31 : key row staged by this thread
  const int sseg = (tid & 7) * 16;    // byte segment within the 1024B row

  for (int key0 = 0; key0 < N_TOK; key0 += BC) {
    // --- kick off async V-chunk staging (8 x b128 per thread) ---
    {
      const char* gsrc = (const char*)(vb + (size_t)(key0 + skey) * C_DIM) + sseg;
      unsigned    ldst = svbase + (unsigned)(skey * SV_LD * 2) + sseg;
#pragma unroll
      for (int i = 0; i < 8; ++i)
        async_copy_b128(ldst + i * 128, gsrc + i * 128);
    }

    // --- S^T tile = K(16 keys x C) @ Q^T(C x 16 rows), pipelined over C ---
    {
      int krow0 = key0 + colg * 16;
      v8f sc = {};
      v16h a   = load_a16(kb, krow0, 0, C_DIM);
      v16h bfr = load_b16(qT, 0, rowg * 16, QT_LD);
      for (int c0 = 0; c0 < C_DIM - 32; c0 += 32) {
        v16h an = load_a16(kb, krow0, c0 + 32, C_DIM);
        v16h bn = load_b16(qT, c0 + 32, rowg * 16, QT_LD);
        sc = wmma_f16(a, bfr, sc);
        a = an; bfr = bn;
      }
      sc = wmma_f16(a, bfr, sc);
      // store transposed back: element (M=key, N=qrow) -> sS[qrow][key]
      int kl = colg * 16 + ((lane >> 4) << 3);
      int rl = rowg * 16 + (lane & 15);
#pragma unroll
      for (int r = 0; r < 8; ++r) sS[rl * S_LD + kl + r] = sc[r];
    }
    __syncthreads();

    // --- online softmax over this 32-key chunk: 4 threads per row ---
    {
      int row = tid >> 2;
      int q4  = tid & 3;
      float sv[8];
      float mloc = -1e30f;
#pragma unroll
      for (int j = 0; j < 8; ++j) {
        sv[j] = sS[row * S_LD + q4 * 8 + j];
        mloc = fmaxf(mloc, sv[j]);
      }
      mloc = fmaxf(mloc, __shfl_xor(mloc, 1, 32));
      mloc = fmaxf(mloc, __shfl_xor(mloc, 2, 32));
      float mold = mrow[row];
      float mnew = fmaxf(mold, mloc);
      float psum = 0.f;
#pragma unroll
      for (int j = 0; j < 8; ++j) {
        float p = __expf(sv[j] - mnew);
        sP[row * P_LD + q4 * 8 + j] = (_Float16)p;
        psum += p;
      }
      psum += __shfl_xor(psum, 1, 32);
      psum += __shfl_xor(psum, 2, 32);
      if (q4 == 0) {
        float fac = __expf(mold - mnew);
        frow[row] = fac;
        lrow[row] = lrow[row] * fac + psum;
        mrow[row] = mnew;
      }
    }
    wait_async0();        // own async copies complete ...
    __syncthreads();      // ... and everyone's are visible

    // --- rescale O, then O += P @ V (one K=32 WMMA step, V from LDS) ---
    {
      int rb = rowg * 16 + ((lane >> 4) << 3);
      float f[8];
#pragma unroll
      for (int r = 0; r < 8; ++r) f[r] = frow[rb + r];
#pragma unroll
      for (int t = 0; t < 16; ++t)
#pragma unroll
        for (int r = 0; r < 8; ++r) acc[t][r] *= f[r];

      v16h a = load_a16(sP, rowg * 16, 0, P_LD);
#pragma unroll
      for (int t = 0; t < 16; ++t) {
        v16h bfr = load_b16(sV, 0, colg * 256 + t * 16, SV_LD);
        acc[t] = wmma_f16(a, bfr, acc[t]);
      }
    }
    __syncthreads();
  }

  // --- normalize by l and write fp16 O for the projection GEMM ---
  {
    int rb = rowg * 16 + ((lane >> 4) << 3);
    int cl = lane & 15;
    float linv[8];
#pragma unroll
    for (int r = 0; r < 8; ++r) linv[r] = 1.f / lrow[rb + r];
    _Float16* ob = o + ((size_t)b * N_TOK + blockIdx.x * BR) * C_DIM;
#pragma unroll
    for (int t = 0; t < 16; ++t) {
      int col = colg * 256 + t * 16 + cl;
#pragma unroll
      for (int r = 0; r < 8; ++r)
        ob[(size_t)(rb + r) * C_DIM + col] = (_Float16)(acc[t][r] * linv[r]);
    }
  }
}

// ---------------- launcher ----------------

extern "C" void kernel_launch(void* const* d_in, const int* in_sizes, int n_in,
                              void* d_out, int out_size, void* d_ws, size_t ws_size,
                              hipStream_t stream) {
  const float* x     = (const float*)d_in[0];
  const float* gamma = (const float*)d_in[1];
  const float* beta  = (const float*)d_in[2];
  const float* wq    = (const float*)d_in[3];
  const float* bq    = (const float*)d_in[4];
  const float* wk    = (const float*)d_in[5];
  const float* bk    = (const float*)d_in[6];
  const float* wv    = (const float*)d_in[7];
  const float* bv    = (const float*)d_in[8];
  const float* wp    = (const float*)d_in[9];
  const float* bp    = (const float*)d_in[10];
  float* out = (float*)d_out;

  char* ws = (char*)d_ws;
  const size_t SZ16 = (size_t)BATCH * N_TOK * C_DIM * sizeof(_Float16);  // 8 MB each
  _Float16* xn16 = (_Float16*)(ws + 0 * SZ16);
  _Float16* q16  = (_Float16*)(ws + 1 * SZ16);
  _Float16* k16  = (_Float16*)(ws + 2 * SZ16);
  _Float16* v16  = (_Float16*)(ws + 3 * SZ16);
  _Float16* o16  = (_Float16*)(ws + 4 * SZ16);
  _Float16* w16  = (_Float16*)(ws + 5 * SZ16);   // 4 x [512x512] fp16 weights
  float2*   stats = (float2*)(ws + 5 * SZ16 + 4 * (size_t)C_DIM * C_DIM * sizeof(_Float16));

  const int WN = C_DIM * C_DIM;
  f32_to_f16_kernel<<<256, 256, 0, stream>>>(wq, w16 + 0 * WN, WN);
  f32_to_f16_kernel<<<256, 256, 0, stream>>>(wk, w16 + 1 * WN, WN);
  f32_to_f16_kernel<<<256, 256, 0, stream>>>(wv, w16 + 2 * WN, WN);
  f32_to_f16_kernel<<<256, 256, 0, stream>>>(wp, w16 + 3 * WN, WN);

  gn_stats_kernel<<<BATCH * GROUPS, 256, 0, stream>>>(x, stats);
  gn_apply_kernel<<<2048, 256, 0, stream>>>(x, gamma, beta, stats, xn16);

  dim3 gg(BATCH * N_TOK / 64, C_DIM / 128);
  float alpha = 1.0f / sqrtf((float)C_DIM);   // fold score scale into Q
  gemm_kernel<0><<<gg, 256, 0, stream>>>(xn16, w16 + 0 * WN, bq, alpha, q16, nullptr, nullptr);
  gemm_kernel<0><<<gg, 256, 0, stream>>>(xn16, w16 + 1 * WN, bk, 1.0f, k16, nullptr, nullptr);
  gemm_kernel<0><<<gg, 256, 0, stream>>>(xn16, w16 + 2 * WN, bv, 1.0f, v16, nullptr, nullptr);

  attention_kernel<<<dim3(N_TOK / BR, BATCH), 256, 0, stream>>>(q16, k16, v16, o16);

  gemm_kernel<1><<<gg, 256, 0, stream>>>(o16, w16 + 3 * WN, bp, 1.0f, nullptr, x, out);
}
